// Net_50397146251564
// MI455X (gfx1250) — matrix operand order repaired and checked by reading, not compiled, not run on hardware
//
#include <hip/hip_runtime.h>
#include <hip/hip_bf16.h>
#include <math.h>

// ---------------------------------------------------------------------------
// Types for CDNA5 WMMA (wave32, 16x16x32 bf16 -> f32)
// ---------------------------------------------------------------------------
typedef __attribute__((ext_vector_type(16))) __bf16 v16bf;
typedef __attribute__((ext_vector_type(2)))  __bf16 v2bf;
typedef __attribute__((ext_vector_type(8)))  float  v8f;

union ABu { uint4 q[2]; unsigned u[8]; unsigned short h[16]; v16bf v; };

// native f32 -> bf16 (RNE via hardware convert); packed pair -> one u32
__device__ __forceinline__ unsigned pack_bf16(float lo, float hi) {
    v2bf p;
    p[0] = (__bf16)lo;
    p[1] = (__bf16)hi;
    return __builtin_bit_cast(unsigned, p);
}

__device__ __forceinline__ unsigned short f2bf(float f) {
    __bf16 h = (__bf16)f;
    return __builtin_bit_cast(unsigned short, h);
}

__device__ __forceinline__ v8f wmma_bf(const ABu& a, const ABu& b, v8f c) {
    return __builtin_amdgcn_wmma_f32_16x16x32_bf16(false, a.v, false, b.v,
                                                   (short)0, c, false, false);
}

template<int NL> struct Mlp {
    const unsigned* W[NL];   // fragment-packed bf16 weights (u32 pairs)
    const float*    bias[NL];// zero-padded to Npad
    int Kp[NL];              // K padded to 32
    int Npad[NL];            // N padded to 32
};

// ---------------------------------------------------------------------------
// Weight conversion: f32 [K,N] row-major -> B-fragment bf16 layout, zero-padded
// u32 index = ((t*(Kp/32)+kk)*256 + L*8 + v); holds (k, k+1) for col n=t*16+(L&15),
// k = kk*32 + ((L<16)?0:8) + 2v + ((v<4)?0:8)  (matches 16-bit B 32x16 WMMA operand layout)
// ---------------------------------------------------------------------------
__global__ void conv_w(const float* __restrict__ W, int K, int N, int Kp, int Npad,
                       unsigned* __restrict__ out) {
    int tot = (Npad / 16) * (Kp / 32) * 256;
    int i = blockIdx.x * 256 + threadIdx.x;
    if (i >= tot) return;
    int per_t = (Kp / 32) * 256;
    int t  = i / per_t;
    int r  = i % per_t;
    int kk = r / 256;
    int e  = r % 256;
    int L  = e / 8;
    int v  = e % 8;
    int n  = t * 16 + (L & 15);
    int k  = kk * 32 + ((L < 16) ? 0 : 8) + 2 * v + ((v < 4) ? 0 : 8);
    float v0 = (k     < K && n < N) ? W[(size_t)k       * N + n] : 0.f;
    float v1 = (k + 1 < K && n < N) ? W[(size_t)(k + 1) * N + n] : 0.f;
    out[i] = pack_bf16(v0, v1);
}

__global__ void conv_b(const float* __restrict__ b, int N, int Npad,
                       float* __restrict__ out) {
    int i = blockIdx.x * 256 + threadIdx.x;
    if (i < Npad) out[i] = (i < N) ? b[i] : 0.f;
}

// zero-fill pad columns of a strided 2D buffer
__global__ void zfill2d(float* __restrict__ dst, int stride, int doff, int cnt,
                        long rows) {
    long i = (long)blockIdx.x * 256 + threadIdx.x;
    long tot = rows * cnt;
    if (i >= tot) return;
    long r = i / cnt;
    int c = (int)(i % cnt);
    dst[r * stride + doff + c] = 0.f;
}

// ---------------------------------------------------------------------------
// Farthest point sampling (matches jnp reference: start at 0, argmax = first max)
// one workgroup per batch
// ---------------------------------------------------------------------------
__global__ void fps_kernel(const float* __restrict__ pos, int Np, int ns,
                           int* __restrict__ idx) {
    int b = blockIdx.x;
    const float* P = pos + (size_t)b * Np * 3;
    __shared__ float sd[4096];
    __shared__ float rv[256];
    __shared__ int   ri[256];
    __shared__ float cur[3];
    int t = threadIdx.x;
    float x0 = P[0], y0 = P[1], z0 = P[2];
    for (int i = t; i < Np; i += 256) {
        float dx = P[i*3+0]-x0, dy = P[i*3+1]-y0, dz = P[i*3+2]-z0;
        sd[i] = dx*dx + dy*dy + dz*dz;
    }
    if (t == 0) idx[b*ns] = 0;
    __syncthreads();
    for (int s = 1; s < ns; s++) {
        float bv = -1.f; int bi = 0x7fffffff;
        for (int i = t; i < Np; i += 256) {
            float v = sd[i];
            if (v > bv || (v == bv && i < bi)) { bv = v; bi = i; }
        }
        rv[t] = bv; ri[t] = bi;
        __syncthreads();
        for (int st = 128; st > 0; st >>= 1) {
            if (t < st) {
                float ov = rv[t+st]; int oi = ri[t+st];
                if (ov > rv[t] || (ov == rv[t] && oi < ri[t])) { rv[t] = ov; ri[t] = oi; }
            }
            __syncthreads();
        }
        int sel = ri[0];
        if (t == 0) {
            idx[b*ns + s] = sel;
            cur[0] = P[sel*3+0]; cur[1] = P[sel*3+1]; cur[2] = P[sel*3+2];
        }
        __syncthreads();
        float cx = cur[0], cy = cur[1], cz = cur[2];
        for (int i = t; i < Np; i += 256) {
            float dx = P[i*3+0]-cx, dy = P[i*3+1]-cy, dz = P[i*3+2]-cz;
            float nd = dx*dx + dy*dy + dz*dz;
            if (nd < sd[i]) sd[i] = nd;
        }
        __syncthreads();
    }
}

__global__ void gather_pos(const float* __restrict__ pos, const int* __restrict__ idx,
                           int Np, int ns, int Bc, float* __restrict__ cen) {
    int i = blockIdx.x * 256 + threadIdx.x;
    if (i >= Bc * ns) return;
    int b = i / ns;
    int j = idx[i];
    const float* p = pos + ((size_t)b * Np + j) * 3;
    cen[i*3+0] = p[0]; cen[i*3+1] = p[1]; cen[i*3+2] = p[2];
}

// ball query: first K(=64) points by index with d2 <= r2 (matches top_k of index keys)
__global__ void ball_query(const float* __restrict__ pos, const float* __restrict__ cen,
                           int Np, int nc, int tot, float r2,
                           int* __restrict__ nb, int* __restrict__ mask) {
    int i = blockIdx.x * 256 + threadIdx.x;
    if (i >= tot) return;
    int b = i / nc;
    const float* P = pos + (size_t)b * Np * 3;
    float cx = cen[i*3+0], cy = cen[i*3+1], cz = cen[i*3+2];
    int cnt = 0;
    for (int j = 0; j < Np && cnt < 64; j++) {
        float dx = P[j*3+0]-cx, dy = P[j*3+1]-cy, dz = P[j*3+2]-cz;
        float d2 = dx*dx + dy*dy + dz*dz;
        if (d2 <= r2) { nb[(size_t)i*64+cnt] = j; mask[(size_t)i*64+cnt] = 1; cnt++; }
    }
    for (; cnt < 64; cnt++) { nb[(size_t)i*64+cnt] = 0; mask[(size_t)i*64+cnt] = 0; }
}

// ---------------------------------------------------------------------------
// Fused SA module: one block (4 wave32) per center. 64 neighbor rows in LDS
// (bf16, ping-pong), full MLP chained via WMMA, masked max in C fragments.
// ---------------------------------------------------------------------------
template<int NL>
__global__ void __launch_bounds__(128)
sa_fused(const float* __restrict__ feat, int featC,
         const float* __restrict__ pos, const float* __restrict__ cen,
         const int* __restrict__ nb, const int* __restrict__ mask,
         int Npts, int nc, Mlp<NL> P, int smax, int NlastReal,
         float* __restrict__ out) {
    extern __shared__ char smraw[];
    unsigned short* bufA = (unsigned short*)smraw;
    unsigned short* bufB = bufA + 64 * smax;
    float* wmax = (float*)(bufB + 64 * smax);   // [4][NpadLast]

    int c = blockIdx.x;
    int b = c / nc;
    int tid = threadIdx.x, w = tid >> 5, lane = tid & 31;
    int Cin = featC + 3;
    float cx = cen[c*3+0], cy = cen[c*3+1], cz = cen[c*3+2];

    // gather + concat -> bf16 LDS input [64][Kp0] (zeros for masked/pad)
    {
        int r = tid >> 1;   // wave w fills its own rows 16w..16w+15
        int j  = nb[(size_t)c*64 + r];
        int mk = mask[(size_t)c*64 + r];
        const float* fj = feat + ((size_t)b * Npts + j) * featC;
        const float* pj = pos  + ((size_t)b * Npts + j) * 3;
        for (int ch = (tid & 1); ch < P.Kp[0]; ch += 2) {
            float v = 0.f;
            if (mk) {
                if (ch < featC) v = fj[ch];
                else if (ch < Cin) {
                    float cc = (ch == featC) ? cx : ((ch == featC+1) ? cy : cz);
                    v = pj[ch - featC] - cc;
                }
            }
            bufA[r * smax + ch] = f2bf(v);
        }
    }
    __syncthreads();

    unsigned short* in  = bufA;
    unsigned short* ob  = bufB;
    int rowBase = 16 * w;
    for (int l = 0; l < NL; l++) {
        int Kp = P.Kp[l], Np = P.Npad[l];
        bool last = (l == NL - 1);
        int m  = rowBase + (lane & 15);
        for (int t = 0; t < Np / 16; t++) {
            float bv = P.bias[l][t*16 + (lane & 15)];
            v8f acc;
            #pragma unroll
            for (int q = 0; q < 8; q++) acc[q] = bv;
            for (int kt = 0; kt < Kp / 32; kt++) {
                int k0 = kt * 32 + ((lane < 16) ? 0 : 8);
                ABu a, bb;
                a.q[0] = *(const uint4*)(in + (size_t)m * smax + k0);
                a.q[1] = *(const uint4*)(in + (size_t)m * smax + k0 + 16);
                const uint4* wp = (const uint4*)(P.W[l] + ((size_t)(t*(Kp/32)+kt)*256 + lane*8));
                bb.q[0] = wp[0]; bb.q[1] = wp[1];
                acc = wmma_bf(a, bb, acc);
            }
            int half8 = (lane < 16) ? 0 : 8;
            if (!last) {
                int n = t*16 + (lane & 15);
                #pragma unroll
                for (int q = 0; q < 8; q++) {
                    float v = acc[q];
                    v = v > 0.f ? v : 0.f;
                    ob[(size_t)(rowBase + half8 + q) * smax + n] = f2bf(v);
                }
            } else {
                float pm = -INFINITY;
                #pragma unroll
                for (int q = 0; q < 8; q++)
                    if (mask[(size_t)c*64 + rowBase + half8 + q]) pm = fmaxf(pm, acc[q]);
                float om = __shfl_xor(pm, 16, 32);
                pm = fmaxf(pm, om);
                if (lane < 16) wmax[w * Np + t*16 + lane] = pm;
            }
        }
        if (!last) { unsigned short* tmp = in; in = ob; ob = tmp; }
    }
    __syncthreads();
    int NpL = P.Npad[NL-1];
    for (int n = tid; n < NlastReal; n += 128) {
        float v = fmaxf(fmaxf(wmax[n], wmax[NpL + n]),
                        fmaxf(wmax[2*NpL + n], wmax[3*NpL + n]));
        out[(size_t)c * NlastReal + n] = v;
    }
}

// ---------------------------------------------------------------------------
// Generic WMMA GEMM (guard-free K): A [M, Astride] f32 with zero-padded columns
// beyond the real K; weights zero-padded to (Kp, Npad). Each wave accumulates
// 4 consecutive 16x16 output tiles, reusing one A fragment per K-step.
// Writes full Npad width (pad cols are exact zeros) with stride Cstride.
// ---------------------------------------------------------------------------
__global__ void __launch_bounds__(128)
gemm_bf16(const float* __restrict__ A, int M, int Astride,
          const unsigned* __restrict__ W, const float* __restrict__ bias,
          int Kp, float* __restrict__ C, int Cstride, int relu) {
    int w = threadIdx.x >> 5, lane = threadIdx.x & 31;
    int rowBase = blockIdx.x * 64 + 16 * w;
    int t0 = blockIdx.y * 4;
    v8f acc[4];
    #pragma unroll
    for (int j = 0; j < 4; j++) {
        float bv = bias[(t0 + j) * 16 + (lane & 15)];
        #pragma unroll
        for (int q = 0; q < 8; q++) acc[j][q] = bv;
    }
    int m  = rowBase + (lane & 15);
    int mc = m < M ? m : (M - 1);
    const float* Ar = A + (size_t)mc * Astride;
    int ktiles = Kp / 32;
    for (int kt = 0; kt < ktiles; kt++) {
        int k0 = kt * 32 + ((lane < 16) ? 0 : 8);
        float4 f0 = *(const float4*)(Ar + k0);
        float4 f1 = *(const float4*)(Ar + k0 + 4);
        float4 f2 = *(const float4*)(Ar + k0 + 16);
        float4 f3 = *(const float4*)(Ar + k0 + 20);
        ABu a;
        a.u[0] = pack_bf16(f0.x, f0.y);
        a.u[1] = pack_bf16(f0.z, f0.w);
        a.u[2] = pack_bf16(f1.x, f1.y);
        a.u[3] = pack_bf16(f1.z, f1.w);
        a.u[4] = pack_bf16(f2.x, f2.y);
        a.u[5] = pack_bf16(f2.z, f2.w);
        a.u[6] = pack_bf16(f3.x, f3.y);
        a.u[7] = pack_bf16(f3.z, f3.w);
        #pragma unroll
        for (int j = 0; j < 4; j++) {
            const uint4* wp = (const uint4*)(W + ((size_t)((t0 + j) * ktiles + kt) * 256 + lane * 8));
            ABu bb;
            bb.q[0] = wp[0]; bb.q[1] = wp[1];
            acc[j] = wmma_bf(a, bb, acc[j]);
        }
    }
    int rb2 = rowBase + ((lane < 16) ? 0 : 8);
    #pragma unroll
    for (int j = 0; j < 4; j++) {
        int n = (t0 + j) * 16 + (lane & 15);
        #pragma unroll
        for (int q = 0; q < 8; q++) {
            int mm = rb2 + q;
            if (mm < M) {
                float v = acc[j][q];
                if (relu && v < 0.f) v = 0.f;
                C[(size_t)mm * Cstride + n] = v;
            }
        }
    }
}

// global max over rows per batch
__global__ void maxpool_rows(const float* __restrict__ h, int rows, int C,
                             float* __restrict__ out) {
    int b = blockIdx.x;
    int ch = blockIdx.y * 256 + threadIdx.x;
    if (ch >= C) return;
    const float* p = h + (size_t)b * rows * C + ch;
    float m = p[0];
    for (int i = 1; i < rows; i++) m = fmaxf(m, p[(size_t)i * C]);
    out[(size_t)b * C + ch] = m;
}

// fp3 concat: [820,3072] = [ xg[b] (2048) | x2[r] (1024) ]  (k=1 interp == broadcast)
__global__ void fp3_cat(const float* __restrict__ xg, const float* __restrict__ x2,
                        float* __restrict__ dst, int rows, int N2) {
    long i = (long)blockIdx.x * 256 + threadIdx.x;
    long tot = (long)rows * 3072;
    if (i >= tot) return;
    long r = i / 3072;
    int cc = (int)(i % 3072);
    int b = (int)(r / N2);
    dst[i] = (cc < 2048) ? xg[(size_t)b * 2048 + cc]
                         : x2[(size_t)r * 1024 + (cc - 2048)];
}

// 3-NN inverse-squared-distance interpolation (matches top_k tie order)
__global__ void knn3_interp(const float* __restrict__ xs, const float* __restrict__ ps,
                            const float* __restrict__ pd, int Ns, int Nd, int C, int Bc,
                            float* __restrict__ out, int ostride) {
    int i = blockIdx.x * 256 + threadIdx.x;
    if (i >= Bc * Nd) return;
    int b = i / Nd;
    const float* P = ps + (size_t)b * Ns * 3;
    float dx0 = pd[i*3+0], dy0 = pd[i*3+1], dz0 = pd[i*3+2];
    float d0 = 1e30f, d1 = 1e30f, d2b = 1e30f;
    int i0 = 0, i1 = 0, i2 = 0;
    for (int j = 0; j < Ns; j++) {
        float dx = P[j*3+0]-dx0, dy = P[j*3+1]-dy0, dz = P[j*3+2]-dz0;
        float d = dx*dx + dy*dy + dz*dz;
        if (d < d0)      { d2b = d1; i2 = i1; d1 = d0; i1 = i0; d0 = d; i0 = j; }
        else if (d < d1) { d2b = d1; i2 = i1; d1 = d; i1 = j; }
        else if (d < d2b){ d2b = d; i2 = j; }
    }
    float w0 = 1.f / fmaxf(d0, 1e-16f);
    float w1 = 1.f / fmaxf(d1, 1e-16f);
    float w2 = 1.f / fmaxf(d2b, 1e-16f);
    float ws = w0 + w1 + w2;
    const float* X = xs + (size_t)b * Ns * C;
    float* o = out + (size_t)i * ostride;
    for (int ch = 0; ch < C; ch++)
        o[ch] = (X[(size_t)i0*C+ch]*w0 + X[(size_t)i1*C+ch]*w1 + X[(size_t)i2*C+ch]*w2) / ws;
}

// strided 2D concat-copy
__global__ void copy2d(const float* __restrict__ src, int sstride, int C,
                       float* __restrict__ dst, int dstride, int doff, long rows) {
    long i = (long)blockIdx.x * 256 + threadIdx.x;
    long tot = rows * C;
    if (i >= tot) return;
    long r = i / C;
    int ch = (int)(i % C);
    dst[r * dstride + doff + ch] = src[r * sstride + ch];
}

// ---------------------------------------------------------------------------
// Fused head MLP: 2 points (64 rows of the [.,32,8] reshape) per block,
// 5 layers via WMMA in LDS, transposed store to d_out [pt,3,32].
// ---------------------------------------------------------------------------
__global__ void __launch_bounds__(128)
head_fused(const float* __restrict__ x5, Mlp<5> P, int smax,
           float* __restrict__ out) {
    extern __shared__ char smraw[];
    unsigned short* bufA = (unsigned short*)smraw;
    unsigned short* bufB = bufA + 64 * smax;
    int blk = blockIdx.x;
    int tid = threadIdx.x, w = tid >> 5, lane = tid & 31;
    {
        int r = tid >> 1;
        int p = blk * 2 + (r >> 5);
        int cc = r & 31;
        const float* src = x5 + (size_t)p * 256 + cc * 8;
        for (int ch = (tid & 1); ch < 32; ch += 2)
            bufA[r * smax + ch] = f2bf((ch < 8) ? src[ch] : 0.f);
    }
    __syncthreads();
    unsigned short* in = bufA;
    unsigned short* ob = bufB;
    int rowBase = 16 * w;
    for (int l = 0; l < 5; l++) {
        int Kp = P.Kp[l], Np = P.Npad[l];
        bool last = (l == 4);
        int m = rowBase + (lane & 15);
        for (int t = 0; t < Np / 16; t++) {
            float bv = P.bias[l][t*16 + (lane & 15)];
            v8f acc;
            #pragma unroll
            for (int q = 0; q < 8; q++) acc[q] = bv;
            for (int kt = 0; kt < Kp / 32; kt++) {
                int k0 = kt * 32 + ((lane < 16) ? 0 : 8);
                ABu a, bb;
                a.q[0] = *(const uint4*)(in + (size_t)m * smax + k0);
                a.q[1] = *(const uint4*)(in + (size_t)m * smax + k0 + 16);
                const uint4* wp = (const uint4*)(P.W[l] + ((size_t)(t*(Kp/32)+kt)*256 + lane*8));
                bb.q[0] = wp[0]; bb.q[1] = wp[1];
                acc = wmma_bf(a, bb, acc);
            }
            int half8 = (lane < 16) ? 0 : 8;
            if (!last) {
                int n = t*16 + (lane & 15);
                #pragma unroll
                for (int q = 0; q < 8; q++) {
                    float v = acc[q];
                    v = v > 0.f ? v : 0.f;
                    ob[(size_t)(rowBase + half8 + q) * smax + n] = f2bf(v);
                }
            } else {
                int n = t*16 + (lane & 15);
                if (n < 3) {
                    #pragma unroll
                    for (int q = 0; q < 8; q++) {
                        int mm = rowBase + half8 + q;
                        int p = blk * 2 + (mm >> 5);
                        int cc = mm & 31;
                        out[(size_t)p * 96 + n * 32 + cc] = acc[q];
                    }
                }
            }
        }
        if (!last) { unsigned short* tmp = in; in = ob; ob = tmp; }
    }
}

// ---------------------------------------------------------------------------
// Host orchestration
// ---------------------------------------------------------------------------
static inline int pad32h(int x) { return (x + 31) & ~31; }

extern "C" void kernel_launch(void* const* d_in, const int* in_sizes, int n_in,
                              void* d_out, int out_size, void* d_ws, size_t ws_size,
                              hipStream_t stream) {
    (void)in_sizes; (void)n_in; (void)out_size; (void)ws_size;
    const int Bc = 4, Nc = 4096, N1 = 820, N2 = 205;
    const float R1 = 0.2f, R2 = 0.4f;
    const float* pos = (const float*)d_in[0];

    struct LD { int K, N; };
    static const LD L[26] = {
        {6,64},{64,128},{128,128},{128,256},            // sa1   (0..3)
        {259,512},{512,512},{512,1024},                 // sa2   (4..6)
        {1027,512},{512,1024},{1024,1024},{1024,2048},  // sa3   (7..10)
        {3072,1024},{1024,512},{512,512},               // fp3   (11..13)
        {768,256},{256,128},{128,128},                  // fp2   (14..16)
        {131,256},{256,512},{512,256},{256,256},        // fp1   (17..20)
        {8,32},{32,128},{128,32},{32,8},{8,3}           // head  (21..25)
    };

    size_t off = 0;
    auto alloc = [&](size_t bytes) -> void* {
        off = (off + 255) & ~(size_t)255;
        void* p = (char*)d_ws + off;
        off += bytes;
        return p;
    };

    unsigned* wfrag[26]; float* bpad[26]; int Kp[26], Npad[26];
    for (int i = 0; i < 26; i++) {
        Kp[i] = pad32h(L[i].K); Npad[i] = pad32h(L[i].N);
        size_t nu32 = (size_t)(Npad[i] / 16) * (Kp[i] / 32) * 256;
        wfrag[i] = (unsigned*)alloc(nu32 * 4);
        bpad[i]  = (float*)alloc((size_t)Npad[i] * 4);
    }
    int*   idx1 = (int*)  alloc((size_t)Bc*N1*4);
    float* pos1 = (float*)alloc((size_t)Bc*N1*3*4);
    int*   nb1  = (int*)  alloc((size_t)Bc*N1*64*4);
    int*   mk1  = (int*)  alloc((size_t)Bc*N1*64*4);
    int*   idx2 = (int*)  alloc((size_t)Bc*N2*4);
    float* pos2 = (float*)alloc((size_t)Bc*N2*3*4);
    int*   nb2  = (int*)  alloc((size_t)Bc*N2*64*4);
    int*   mk2  = (int*)  alloc((size_t)Bc*N2*64*4);
    float* x1   = (float*)alloc((size_t)Bc*N1*256*4);
    float* x2   = (float*)alloc((size_t)Bc*N2*1024*4);
    float* cat3 = (float*)alloc((size_t)Bc*N2*1056*4);   // padded stride 1056
    float* gA   = (float*)alloc((size_t)Bc*N2*2048*4);   // ping (>= 3280*256 too)
    float* gB   = (float*)alloc((size_t)Bc*N2*2048*4);   // pong
    float* xg   = (float*)alloc((size_t)Bc*2048*4);
    float* cat33= (float*)alloc((size_t)Bc*N2*3072*4);
    float* x3   = (float*)alloc((size_t)Bc*N2*512*4);
    float* catf2= (float*)alloc((size_t)Bc*N1*768*4);
    float* x4   = (float*)alloc((size_t)Bc*N1*128*4);
    float* catf1= (float*)alloc((size_t)Bc*Nc*160*4);    // padded stride 160
    float* p1   = (float*)alloc((size_t)Bc*Nc*512*4);
    float* p2   = (float*)alloc((size_t)Bc*Nc*512*4);

    // weight / bias packing
    for (int i = 0; i < 26; i++) {
        const float* W = (const float*)d_in[1 + 2*i];
        const float* b = (const float*)d_in[2 + 2*i];
        int nu32 = (Npad[i] / 16) * (Kp[i] / 32) * 256;
        conv_w<<<dim3((nu32 + 255) / 256), dim3(256), 0, stream>>>(W, L[i].K, L[i].N, Kp[i], Npad[i], wfrag[i]);
        conv_b<<<dim3((Npad[i] + 255) / 256), dim3(256), 0, stream>>>(b, L[i].N, Npad[i], bpad[i]);
    }

    // guard-free GEMM: A stride is padded to Kp with zeroed pad columns
    auto gemm = [&](const float* A, int M, int Astride, int li, float* C, int relu) {
        dim3 g((M + 63) / 64, Npad[li] / 64);
        gemm_bf16<<<g, dim3(128), 0, stream>>>(A, M, Astride, wfrag[li], bpad[li],
                                               Kp[li], C, Npad[li], relu);
    };

    // ---- SA1 ----
    fps_kernel<<<dim3(Bc), dim3(256), 0, stream>>>(pos, Nc, N1, idx1);
    gather_pos<<<dim3((Bc*N1 + 255)/256), dim3(256), 0, stream>>>(pos, idx1, Nc, N1, Bc, pos1);
    ball_query<<<dim3((Bc*N1 + 255)/256), dim3(256), 0, stream>>>(pos, pos1, Nc, N1, Bc*N1, R1*R1, nb1, mk1);
    {
        Mlp<4> P;
        for (int l = 0; l < 4; l++) { P.W[l]=wfrag[l]; P.bias[l]=bpad[l]; P.Kp[l]=Kp[l]; P.Npad[l]=Npad[l]; }
        int smax = 256;
        size_t sh = (size_t)2*64*smax*2 + (size_t)4*Npad[3]*4;
        sa_fused<4><<<dim3(Bc*N1), dim3(128), sh, stream>>>(pos, 3, pos, pos1, nb1, mk1, Nc, N1, P, smax, 256, x1);
    }
    // ---- SA2 ----
    fps_kernel<<<dim3(Bc), dim3(256), 0, stream>>>(pos1, N1, N2, idx2);
    gather_pos<<<dim3((Bc*N2 + 255)/256), dim3(256), 0, stream>>>(pos1, idx2, N1, N2, Bc, pos2);
    ball_query<<<dim3((Bc*N2 + 255)/256), dim3(256), 0, stream>>>(pos1, pos2, N1, N2, Bc*N2, R2*R2, nb2, mk2);
    {
        Mlp<3> P;
        for (int l = 0; l < 3; l++) { P.W[l]=wfrag[4+l]; P.bias[l]=bpad[4+l]; P.Kp[l]=Kp[4+l]; P.Npad[l]=Npad[4+l]; }
        int smax = 1024;
        size_t sh = (size_t)2*64*smax*2 + (size_t)4*Npad[6]*4;
        sa_fused<3><<<dim3(Bc*N2), dim3(128), sh, stream>>>(x1, 256, pos1, pos2, nb2, mk2, N1, N2, P, smax, 1024, x2);
    }
    // ---- Global SA (sa3) ----  cat3 stride 1056, pad cols 1027..1055 zeroed
    copy2d<<<dim3((unsigned)(((long)Bc*N2*1024 + 255)/256)), dim3(256), 0, stream>>>(x2, 1024, 1024, cat3, 1056, 0, (long)Bc*N2);
    copy2d<<<dim3((unsigned)(((long)Bc*N2*3 + 255)/256)),    dim3(256), 0, stream>>>(pos2, 3, 3, cat3, 1056, 1024, (long)Bc*N2);
    zfill2d<<<dim3((unsigned)(((long)Bc*N2*29 + 255)/256)),  dim3(256), 0, stream>>>(cat3, 1056, 1027, 29, (long)Bc*N2);
    gemm(cat3, Bc*N2, 1056, 7,  gA, 1);
    gemm(gA,   Bc*N2, 512,  8,  gB, 1);
    gemm(gB,   Bc*N2, 1024, 9,  gA, 1);
    gemm(gA,   Bc*N2, 1024, 10, gB, 0);
    maxpool_rows<<<dim3(Bc, (2048 + 255)/256), dim3(256), 0, stream>>>(gB, N2, 2048, xg);
    // ---- FP3 (k=1 interp == broadcast of xg) ----
    fp3_cat<<<dim3((unsigned)(((long)Bc*N2*3072 + 255)/256)), dim3(256), 0, stream>>>(xg, x2, cat33, Bc*N2, N2);
    gemm(cat33, Bc*N2, 3072, 11, gA, 1);
    gemm(gA,    Bc*N2, 1024, 12, gB, 1);
    gemm(gB,    Bc*N2, 512,  13, x3, 0);
    // ---- FP2 ----
    knn3_interp<<<dim3((Bc*N1 + 255)/256), dim3(256), 0, stream>>>(x3, pos2, pos1, N2, N1, 512, Bc, catf2, 768);
    copy2d<<<dim3((unsigned)(((long)Bc*N1*256 + 255)/256)), dim3(256), 0, stream>>>(x1, 256, 256, catf2, 768, 512, (long)Bc*N1);
    gemm(catf2, Bc*N1, 768, 14, gA, 1);
    gemm(gA,    Bc*N1, 256, 15, gB, 1);
    gemm(gB,    Bc*N1, 128, 16, x4, 0);
    // ---- FP1 ----  catf1 stride 160, pad cols 131..159 zeroed
    knn3_interp<<<dim3((Bc*Nc + 255)/256), dim3(256), 0, stream>>>(x4, pos1, pos, N1, Nc, 128, Bc, catf1, 160);
    copy2d<<<dim3((unsigned)(((long)Bc*Nc*3 + 255)/256)), dim3(256), 0, stream>>>(pos, 3, 3, catf1, 160, 128, (long)Bc*Nc);
    zfill2d<<<dim3((unsigned)(((long)Bc*Nc*29 + 255)/256)), dim3(256), 0, stream>>>(catf1, 160, 131, 29, (long)Bc*Nc);
    gemm(catf1, Bc*Nc, 160, 17, p1, 1);
    gemm(p1,    Bc*Nc, 256, 18, p2, 1);
    gemm(p2,    Bc*Nc, 512, 19, p1, 1);
    gemm(p1,    Bc*Nc, 256, 20, p2, 0);
    // ---- Head (fused, transposed store) ----
    {
        Mlp<5> P;
        for (int l = 0; l < 5; l++) { P.W[l]=wfrag[21+l]; P.bias[l]=bpad[21+l]; P.Kp[l]=Kp[21+l]; P.Npad[l]=Npad[21+l]; }
        int smax = 128;
        size_t sh = (size_t)2*64*smax*2;
        head_fused<<<dim3(Bc*Nc/2), dim3(128), sh, stream>>>(p2, P, smax, (float*)d_out);
    }
}
